// DCLLoss_72172630442628
// MI455X (gfx1250) — matrix operand order repaired and checked by reading, not compiled, not run on hardware
//
#include <hip/hip_runtime.h>
#include <hip/hip_bf16.h>

// ---------------------------------------------------------------------------
// DCL contrastive loss, fused on CDNA5 (gfx1250).
//   loss = -mean(diag(img·molT/T)) + 0.5*mean(LSE_A+LSE_B+LSE_C+LSE_D)
//   A=img·imgT, B=mol·molT, C=img·molT, D=mol·imgT, all /T, diag masked.
// Logits bounded in [-10,10] => plain sum-of-exp in fp32 (no max pass).
//
// Data path: bf16 operands pre-scaled by sqrt(10*log2 e) so WMMA outputs are
// directly exp2 arguments (no per-element scaling VALU); v_wmma_f32_16x16x32_bf16;
// B tiles double-buffer-staged into LDS with global_load_async_to_lds_b128
// (ASYNCcnt) shared by all 8 waves; each wave holds 2 row tiles of A (K=128)
// in registers => 8 WMMAs per staged B tile read, ~256 MB total global traffic.
// ---------------------------------------------------------------------------

typedef __attribute__((ext_vector_type(16))) __bf16 v16bf;
typedef __attribute__((ext_vector_type(8)))  __bf16 v8bf;
typedef __attribute__((ext_vector_type(8)))  float  v8f;

#define NROWS 8192
#define CDIM  128
// dot(x_scaled, y_scaled) = (x·y) * 10 * log2(e)  ==  logit * log2(e)
#define PRESCALE 3.7982826203717246f          // sqrt(10 * log2(e))
#define LN2      0.6931471805599453f

// stage = 4 column tiles = 64 Y rows = 16 KB (contiguous, Y is row-major)
#define TILES_PER_STAGE 4
#define STAGE_BYTES     (TILES_PER_STAGE * 16 * CDIM * 2)   // 16384
#define NSTAGES         (NROWS / (TILES_PER_STAGE * 16))    // 128

// ---------------------------------------------------------------------------
// Kernel 1: L2-normalize rows (fp32, matches reference), fold in PRESCALE,
// round to bf16.
// ---------------------------------------------------------------------------
__global__ void dcl_normalize_kernel(const float* __restrict__ img,
                                     const float* __restrict__ mol,
                                     __hip_bfloat16* __restrict__ imgb,
                                     __hip_bfloat16* __restrict__ molb) {
    int row = blockIdx.x;                 // 0 .. 2*NROWS-1
    const float* src;
    __hip_bfloat16* dst;
    if (row < NROWS) { src = img + (size_t)row * CDIM;          dst = imgb + (size_t)row * CDIM; }
    else             { int r = row - NROWS;
                       src = mol + (size_t)r * CDIM;            dst = molb + (size_t)r * CDIM; }
    int t = threadIdx.x;                  // CDIM = 128 threads (4 waves)
    float x  = src[t];
    float ss = x * x;
    for (int off = 1; off < 32; off <<= 1) ss += __shfl_xor(ss, off, 32);
    __shared__ float red[4];
    int wid = t >> 5, lane = t & 31;
    if (lane == 0) red[wid] = ss;
    __syncthreads();
    float tot = red[0] + red[1] + red[2] + red[3];
    float inv = (1.0f / fmaxf(sqrtf(tot), 1e-12f)) * PRESCALE;
    dst[t] = __float2bfloat16(x * inv);
}

// ---------------------------------------------------------------------------
// Build a 16x16x32 bf16 WMMA operand from two 8-element contiguous chunks.
// ---------------------------------------------------------------------------
static __device__ inline v16bf pack16(const __bf16* p, int o0, int o1) {
    v8bf lo = *reinterpret_cast<const v8bf*>(p + o0);   // 16B aligned
    v8bf hi = *reinterpret_cast<const v8bf*>(p + o1);
    v16bf v;
#pragma unroll
    for (int i = 0; i < 8; ++i) { v[i] = lo[i]; v[8 + i] = hi[i]; }
    return v;
}

// ---------------------------------------------------------------------------
// Kernel 2: fused GEMM + per-row sum(exp2(logit*log2e)).
// Jobs: 0:A=img·imgT 1:B=mol·molT 2:C=img·molT 3:D=mol·imgT
// Block = 256 thr = 8 waves; wave owns 2 row tiles (32 rows) => 256 rows/block.
// grid = 4 jobs * 32 blocks. B tiles double-buffer-staged in LDS via async DMA.
// ---------------------------------------------------------------------------
__global__ __launch_bounds__(256) void dcl_sim_lse_kernel(
        const __hip_bfloat16* __restrict__ imgb,
        const __hip_bfloat16* __restrict__ molb,
        float* __restrict__ lse,      // [4 * NROWS]
        float* __restrict__ diagArr)  // [NROWS] (natural logits of diag(C))
{
    __shared__ __align__(16) __bf16 sbuf[2][STAGE_BYTES / 2];

    const int job = blockIdx.x >> 5;          // 0..3
    const int rb  = blockIdx.x & 31;          // row-block within job
    const __hip_bfloat16 *Xp, *Yp;
    if      (job == 0) { Xp = imgb; Yp = imgb; }
    else if (job == 1) { Xp = molb; Yp = molb; }
    else if (job == 2) { Xp = imgb; Yp = molb; }
    else               { Xp = molb; Yp = imgb; }
    const __bf16* X = reinterpret_cast<const __bf16*>(Xp);
    const __bf16* Y = reinterpret_cast<const __bf16*>(Yp);

    const int tid  = threadIdx.x;
    const int lane = tid & 31;
    const int wave = tid >> 5;
    const int hlf  = lane >> 4;               // lane half (0/1)
    const int l16  = lane & 15;
    const int rowbase = rb * 256 + wave * 32; // this wave's 32 rows (2 tiles)

    // --- A operand: 2 row tiles x 4 K-chunks held in registers for the loop.
    // A 16x32 layout: lanes 0-15 hold K 0..15, lanes 16-31 hold K 16..31 (matches B).
    v16bf a[2][4];
#pragma unroll
    for (int t = 0; t < 2; ++t) {
        const __bf16* xr = X + (size_t)(rowbase + t * 16 + l16) * CDIM;
#pragma unroll
        for (int kk = 0; kk < 4; ++kk)
            a[t][kk] = pack16(xr, kk * 32 + (hlf ? 16 : 0), kk * 32 + (hlf ? 24 : 8));
    }

    // --- scalar diagonal "log2-domain logit" for row rowbase+lane
    float dlog2;
    {
        const __bf16* xr = X + (size_t)(rowbase + lane) * CDIM;
        const __bf16* yr = Y + (size_t)(rowbase + lane) * CDIM;
        float s = 0.f;
#pragma unroll 8
        for (int i = 0; i < CDIM; ++i) s += (float)xr[i] * (float)yr[i];
        dlog2 = s;                            // = logit * log2(e)
    }

    float sums[2][8];
#pragma unroll
    for (int t = 0; t < 2; ++t)
#pragma unroll
        for (int r = 0; r < 8; ++r) sums[t][r] = 0.f;

    // LDS byte offsets of the two stage buffers (generic ptr low 32 bits = LDS offset)
    const unsigned ldsBase0 = (unsigned)(size_t)(&sbuf[0][0]);
    const unsigned ldsBase1 = (unsigned)(size_t)(&sbuf[1][0]);

    // issue one 16 KB stage: 256 threads x 4 async b128 chunks
    auto stage_issue = [&](int s, unsigned ldsBase) {
        const char* g = (const char*)Y + (size_t)s * STAGE_BYTES;
#pragma unroll
        for (int k = 0; k < 4; ++k) {
            unsigned long long ga = (unsigned long long)(size_t)(g + tid * 16 + k * 4096);
            unsigned la = ldsBase + (unsigned)(tid * 16 + k * 4096);
            asm volatile("global_load_async_to_lds_b128 %0, %1, off"
                         :: "v"(la), "v"(ga) : "memory");
        }
    };

    stage_issue(0, ldsBase0);

    for (int s = 0; s < NSTAGES; ++s) {
        // prefetch next stage into the other buffer, then wait for this stage
        if (s + 1 < NSTAGES) {
            stage_issue(s + 1, ((s + 1) & 1) ? ldsBase1 : ldsBase0);
            asm volatile("s_wait_asynccnt 0x4" ::: "memory");   // next stage may stay in flight
        } else {
            asm volatile("s_wait_asynccnt 0x0" ::: "memory");
        }
        __syncthreads();                    // stage s visible to all waves

        const __bf16* buf = &sbuf[s & 1][0];
#pragma unroll
        for (int ct = 0; ct < TILES_PER_STAGE; ++ct) {
            // B 32x16 layout: contiguous 16 K values per lane-half (K 0-15 / 16-31)
            const __bf16* yr = buf + (size_t)(ct * 16 + l16) * CDIM;
            v16bf b[4];
#pragma unroll
            for (int kk = 0; kk < 4; ++kk) {
                int o0 = kk * 32 + (hlf ? 16 : 0);
                b[kk] = pack16(yr, o0, o0 + 8);
            }
#pragma unroll
            for (int t = 0; t < 2; ++t) {
                v8f d = {0.f, 0.f, 0.f, 0.f, 0.f, 0.f, 0.f, 0.f};
#pragma unroll
                for (int kk = 0; kk < 4; ++kk)
                    d = __builtin_amdgcn_wmma_f32_16x16x32_bf16(
                            false, a[t][kk], false, b[kk], (short)0, d, false, false);
                // D layout: lane holds column N=l16, rows r+8*hlf in d[r];
                // d[r] is already logit*log2(e) thanks to PRESCALE.
#pragma unroll
                for (int r = 0; r < 8; ++r)
                    sums[t][r] += __builtin_amdgcn_exp2f(d[r]);
            }
        }
        __syncthreads();                    // all waves done reading buf[s&1]
    }

    // --- reduce across the 16 lanes that share each row (butterfly in half-wave)
#pragma unroll
    for (int t = 0; t < 2; ++t)
#pragma unroll
        for (int r = 0; r < 8; ++r) {
            float v = sums[t][r];
            v += __shfl_xor(v, 1, 32);
            v += __shfl_xor(v, 2, 32);
            v += __shfl_xor(v, 4, 32);
            v += __shfl_xor(v, 8, 32);
            sums[t][r] = v;
        }

    // writer lanes: 0..7 -> rows 0..7 of tile, 16..23 -> rows 8..15
    const int mw = (lane & 7) + hlf * 8;
#pragma unroll
    for (int t = 0; t < 2; ++t) {
        float dv = __shfl(dlog2, t * 16 + mw, 32);      // log2-domain diag logit
        if ((lane & 15) < 8) {
            float S = sums[t][lane & 7] - __builtin_amdgcn_exp2f(dv);  // mask diagonal
            int row = rowbase + t * 16 + mw;
            lse[job * NROWS + row] = logf(fmaxf(S, 1e-30f));
            if (job == 2) diagArr[row] = dv * LN2;      // natural-units positive logit
        }
    }
}

// ---------------------------------------------------------------------------
// Kernel 3: deterministic final reduction to the scalar loss.
// ---------------------------------------------------------------------------
__global__ void dcl_finalize_kernel(const float* __restrict__ lse,
                                    const float* __restrict__ diagArr,
                                    float* __restrict__ out) {
    __shared__ float sh[256];
    int t = threadIdx.x;
    float accL = 0.f, accD = 0.f;
    for (int i = t; i < 4 * NROWS; i += 256) accL += lse[i];
    for (int i = t; i < NROWS; i += 256)     accD += diagArr[i];
    sh[t] = 0.5f * accL - accD;
    __syncthreads();
    for (int s = 128; s > 0; s >>= 1) {
        if (t < s) sh[t] += sh[t + s];
        __syncthreads();
    }
    if (t == 0) out[0] = sh[0] / (float)NROWS;
}

// ---------------------------------------------------------------------------
extern "C" void kernel_launch(void* const* d_in, const int* in_sizes, int n_in,
                              void* d_out, int out_size, void* d_ws, size_t ws_size,
                              hipStream_t stream) {
    (void)in_sizes; (void)n_in; (void)out_size; (void)ws_size;
    const float* img = (const float*)d_in[0];
    const float* mol = (const float*)d_in[1];

    char* ws = (char*)d_ws;
    __hip_bfloat16* imgb = (__hip_bfloat16*)ws;                                     // 2 MB
    __hip_bfloat16* molb = (__hip_bfloat16*)(ws + (size_t)NROWS * CDIM * 2);        // 2 MB
    float* lse   = (float*)(ws + (size_t)NROWS * CDIM * 4);                         // 128 KB
    float* diagA = (float*)(ws + (size_t)NROWS * CDIM * 4 + (size_t)4 * NROWS * 4); // 32 KB

    dcl_normalize_kernel<<<2 * NROWS, CDIM, 0, stream>>>(img, mol, imgb, molb);
    dcl_sim_lse_kernel<<<4 * 32, 256, 0, stream>>>(imgb, molb, lse, diagA);
    dcl_finalize_kernel<<<1, 256, 0, stream>>>(lse, diagA, (float*)d_out);
}
